// AttendedSeqEmbedding_56461640073946
// MI455X (gfx1250) — matrix operand order repaired and checked by reading, not compiled, not run on hardware
//
#include <hip/hip_runtime.h>

// ---------------------------------------------------------------------------
// Bidirectional GRU + MLP + attention pooling for MI455X (gfx1250, wave32).
//
// 16 sequences per 128-thread block (4 waves). Each timestep:
//   gx = x_t * WihT   (16x128x192, f16 WMMA, f32 acc)
//   gh = h   * WhhT   (16x64x192)
//   GRU elementwise update (fp32, masked freeze past seqlen)
//   fwd pass only: o = tanh([hf|hb] * WmT + bm), online-softmax attention.
// Backward hidden states spilled as f16 (33.5 MB, L2-resident: L2 = 192 MB).
// Per-step staging of x / h_b uses GLOBAL_LOAD_ASYNC_TO_LDS_B128 (ASYNCcnt).
// Weight B-fragments are register-resident (1 wave/SIMD regime -> VGPRs free).
// ---------------------------------------------------------------------------

#define DEV __device__ __forceinline__

typedef _Float16 f16;
typedef __attribute__((ext_vector_type(16))) _Float16 v16h;
typedef __attribute__((ext_vector_type(8)))  float    v8f;

static constexpr int Bsz  = 512;  // batch (number of sequences)
static constexpr int Tmax = 512;  // max seq length
static constexpr int Idim = 128;  // input dim
static constexpr int Hdim = 64;   // hidden dim
static constexpr int Gdim = 192;  // 3*H gate dim
static constexpr int Odim = 128;  // output / MLP dim

// --- WMMA fragment maps (wave32, 16-bit operands) ---------------------------
// A matrix 16x32 (f16), per CDNA5 ISA 7.12.2:
//   lane = m + 16*half ; VGPR j holds K = (j%4)*2+p + half*8 + (j/4)*16
DEV int amap(int m, int k) {
  int t16  = k >> 4;
  int rem  = k & 15;
  int half = rem >> 3;
  int rem8 = rem & 7;
  int vj   = (rem8 >> 1) + 4 * t16;
  int p    = rem8 & 1;
  return (m + 16 * half) * 16 + (2 * vj + p);
}

DEV v8f wmma_f16(v16h a, v16h b, v8f c) {
  return __builtin_amdgcn_wmma_f32_16x16x32_f16(false, a, false, b,
                                                (short)0, c, false, false);
}

DEV float sigmoidf_(float x) { return 1.0f / (1.0f + __expf(-x)); }

// --- gfx1250 async global->LDS copy (16B per lane, tracked by ASYNCcnt) -----
// GV addressing form: LDS byte address taken from the vdst VGPR, global address from the vaddr VGPR pair.
// Low 32 bits of a generic pointer to __shared__ are the LDS byte offset
// (ISA 10.2 aperture mapping).
DEV void async_ld_b128(void* lds_dst, const void* gsrc) {
  unsigned lds = (unsigned)(unsigned long long)lds_dst;
  unsigned long long ga = (unsigned long long)gsrc;
  asm volatile("global_load_async_to_lds_b128 %0, %1, off"
               :: "v"(lds), "v"(ga) : "memory");
}
DEV void async_wait0() {
  asm volatile("s_wait_asynccnt 0" ::: "memory");
}

// --- kernel: exclusive prefix sum of seqlens -> starts ----------------------
__global__ void k_starts(const int* __restrict__ lens, int* __restrict__ starts) {
  if (threadIdx.x == 0 && blockIdx.x == 0) {
    int acc = 0;
    for (int b = 0; b < Bsz; b++) { starts[b] = acc; acc += lens[b]; }
  }
}

// --- kernel: pack fp32 weight [N][K] into f16 B-fragment layout -------------
// B matrix 32x16: lane half h, element e holds K = 8h + (e&7) + 16*(e>>3).
__global__ void k_pack(const float* __restrict__ W, f16* __restrict__ dst,
                       int N, int K) {
  int idx = blockIdx.x * 256 + threadIdx.x;
  if (idx >= N * K) return;
  int numC   = K >> 5;
  int frag   = idx >> 9;
  int within = idx & 511;
  int lane   = within >> 4;
  int e      = within & 15;
  int nt     = frag / numC;
  int c      = frag - nt * numC;
  int k      = 32 * c + ((lane >> 4) << 3) + (e & 7) + ((e >> 3) << 4);
  int n      = 16 * nt + (lane & 15);
  dst[idx] = (f16)W[(size_t)n * K + k];
}

// --- kernel: backward-direction GRU scan, writes h_b (f16) at original t ----
__global__ __launch_bounds__(128) void k_bwd(
    const float* __restrict__ seqs, const int* __restrict__ starts,
    const int* __restrict__ lens,
    const f16* __restrict__ wih, const f16* __restrict__ whh,
    const float* __restrict__ bih, const float* __restrict__ bhh,
    f16* __restrict__ hb_out) {
  __shared__ float s_agx[16 * Gdim];
  __shared__ float s_agh[16 * Gdim];
  __shared__ __align__(16) float s_xrow[16 * Idim];
  __shared__ __align__(32) f16 s_xfrag[4 * 512];
  __shared__ __align__(32) f16 s_hfrag[2 * 512];
  __shared__ float s_hcur[16 * Hdim];
  __shared__ float s_bih[Gdim], s_bhh[Gdim];
  __shared__ int s_start[16], s_len[16], s_kmax;

  const int tid  = threadIdx.x;
  const int wave = tid >> 5;
  const int lane = tid & 31;
  const int b0   = blockIdx.x * 16;

  if (tid < 16) { s_start[tid] = starts[b0 + tid]; s_len[tid] = lens[b0 + tid]; }
  for (int i = tid; i < Gdim; i += 128) { s_bih[i] = bih[i]; s_bhh[i] = bhh[i]; }
  for (int idx = tid; idx < 16 * Hdim; idx += 128) {
    int m = idx >> 6, j = idx & 63;
    s_hcur[idx] = 0.0f;
    s_hfrag[(j >> 5) * 512 + amap(m, j & 31)] = (f16)0.0f;
  }
  __syncthreads();
  if (tid == 0) {
    int km = 0;
    for (int m = 0; m < 16; m++) km = (s_len[m] > km) ? s_len[m] : km;
    s_kmax = km;
  }
  __syncthreads();
  const int kmax = s_kmax;

  // register-resident weight B-fragments (loop-invariant)
  v16h wihr[3][4], whf[3][2];
  for (int i = 0; i < 3; i++) {
    for (int c = 0; c < 4; c++)
      wihr[i][c] = *(const v16h*)(wih + (((3 * wave + i) * 4 + c) * 32 + lane) * 16);
    for (int c = 0; c < 2; c++)
      whf[i][c] = *(const v16h*)(whh + (((3 * wave + i) * 2 + c) * 32 + lane) * 16);
  }

  for (int k = 0; k < kmax; k++) {
    // stage x_t rows (time-reversed) via async global->LDS (16B/lane x 4)
    for (int q = 0; q < 4; q++) {
      int fo  = (tid + 128 * q) * 4;       // float offset in [0, 2048)
      int m   = fo >> 7, col = fo & 127;
      int t   = s_len[m] - 1 - k; t = (t < 0) ? 0 : t;
      const float* g = seqs + (size_t)(s_start[m] + t) * Idim + col;
      async_ld_b128(&s_xrow[fo], g);
    }
    { // prefetch next step's rows while GEMMs run
      int m = tid >> 3;
      int t2 = s_len[m] - 2 - k; t2 = (t2 < 0) ? 0 : t2;
      __builtin_prefetch(seqs + (size_t)(s_start[m] + t2) * Idim + (tid & 7) * 16, 0, 0);
    }
    async_wait0();
    __syncthreads();
    for (int r = 0; r < 16; r++) { // pack x -> A fragments (f16)
      int idx = tid + 128 * r;
      int m = idx >> 7, j = idx & 127;
      s_xfrag[(j >> 5) * 512 + amap(m, j & 31)] = (f16)s_xrow[idx];
    }
    __syncthreads();
    // A fragments once per step, reused across this wave's 3 N-tiles
    v16h ax[4], ah[2];
    for (int c = 0; c < 4; c++) ax[c] = *(const v16h*)(s_xfrag + c * 512 + lane * 16);
    for (int c = 0; c < 2; c++) ah[c] = *(const v16h*)(s_hfrag + c * 512 + lane * 16);
    for (int i = 0; i < 3; i++) {
      int tg = 3 * wave + i;
      v8f cg = {}, ch = {};
      for (int c = 0; c < 4; c++) cg = wmma_f16(ax[c], wihr[i][c], cg);
      for (int c = 0; c < 2; c++) ch = wmma_f16(ah[c], whf[i][c], ch);
      int n = tg * 16 + (lane & 15);
      int hh = lane >> 4;
      for (int r = 0; r < 8; r++) {
        int m = r + 8 * hh;
        s_agx[m * Gdim + n] = cg[r];
        s_agh[m * Gdim + n] = ch[r];
      }
    }
    __syncthreads();
    // element-wise GRU update (mask freezes h past seq end)
    for (int r = 0; r < 8; r++) {
      int idx = tid + 128 * r;
      int m = idx >> 6, j = idx & 63;
      float ar  = (s_agx[m*Gdim + j]      + s_bih[j])      + (s_agh[m*Gdim + j]      + s_bhh[j]);
      float az  = (s_agx[m*Gdim + 64 + j] + s_bih[64 + j]) + (s_agh[m*Gdim + 64 + j] + s_bhh[64 + j]);
      float gxn =  s_agx[m*Gdim + 128 + j] + s_bih[128 + j];
      float ghn =  s_agh[m*Gdim + 128 + j] + s_bhh[128 + j];
      float rg = sigmoidf_(ar);
      float zg = sigmoidf_(az);
      float ng = tanhf(gxn + rg * ghn);
      float hold = s_hcur[idx];
      float hv = (k < s_len[m]) ? ((1.0f - zg) * ng + zg * hold) : hold;
      s_hcur[idx] = hv;
      s_hfrag[(j >> 5) * 512 + amap(m, j & 31)] = (f16)hv;
      if (k < s_len[m]) {
        int t = s_len[m] - 1 - k;
        hb_out[((size_t)(b0 + m) * Tmax + t) * Hdim + j] = (f16)hv;
      }
    }
    __syncthreads();
  }
}

// --- kernel: forward scan + concat(hb) + MLP + online-softmax attention -----
__global__ __launch_bounds__(128) void k_fwd(
    const float* __restrict__ seqs, const int* __restrict__ starts,
    const int* __restrict__ lens,
    const f16* __restrict__ wih, const f16* __restrict__ whh,
    const f16* __restrict__ wm,
    const float* __restrict__ bih, const float* __restrict__ bhh,
    const float* __restrict__ bm, const float* __restrict__ ctx,
    const f16* __restrict__ hb, float* __restrict__ out) {
  __shared__ float s_agx[16 * Gdim];
  __shared__ float s_agh[16 * Gdim];
  __shared__ __align__(16) float s_xrow[16 * Idim];
  __shared__ __align__(32) f16 s_xfrag[4 * 512];
  __shared__ __align__(32) f16 s_hfrag[2 * 512];
  __shared__ __align__(32) f16 s_mfrag[4 * 512];   // [hf | hb] K=128 A-fragments
  __shared__ __align__(16) f16 s_hbrow[16 * Hdim];
  __shared__ float s_hcur[16 * Hdim];
  __shared__ float s_bih[Gdim], s_bhh[Gdim], s_bm[Odim], s_ctx[Odim];
  __shared__ float s_score[16], s_maxv[16], s_sumv[16], s_alpha[16], s_w[16];
  __shared__ int s_start[16], s_len[16], s_kmax;

  const int tid  = threadIdx.x;
  const int wave = tid >> 5;
  const int lane = tid & 31;
  const int b0   = blockIdx.x * 16;

  if (tid < 16) {
    s_start[tid] = starts[b0 + tid];
    s_len[tid]   = lens[b0 + tid];
    s_score[tid] = 0.0f;
    s_maxv[tid]  = -3.0e38f;
    s_sumv[tid]  = 0.0f;
  }
  for (int i = tid; i < Gdim; i += 128) { s_bih[i] = bih[i]; s_bhh[i] = bhh[i]; }
  for (int i = tid; i < Odim; i += 128) { s_bm[i] = bm[i]; s_ctx[i] = ctx[i]; }
  for (int idx = tid; idx < 16 * Hdim; idx += 128) {
    int m = idx >> 6, j = idx & 63;
    s_hcur[idx] = 0.0f;
    s_hfrag[(j >> 5) * 512 + amap(m, j & 31)] = (f16)0.0f;
  }
  __syncthreads();
  if (tid == 0) {
    int km = 0;
    for (int m = 0; m < 16; m++) km = (s_len[m] > km) ? s_len[m] : km;
    s_kmax = km;
  }
  __syncthreads();
  const int kmax = s_kmax;

  // register-resident weight B-fragments (loop-invariant)
  v16h wihr[3][4], whf[3][2], wmf[2][4];
  for (int i = 0; i < 3; i++) {
    for (int c = 0; c < 4; c++)
      wihr[i][c] = *(const v16h*)(wih + (((3 * wave + i) * 4 + c) * 32 + lane) * 16);
    for (int c = 0; c < 2; c++)
      whf[i][c] = *(const v16h*)(whh + (((3 * wave + i) * 2 + c) * 32 + lane) * 16);
  }
  for (int i = 0; i < 2; i++)
    for (int c = 0; c < 4; c++)
      wmf[i][c] = *(const v16h*)(wm + (((2 * wave + i) * 4 + c) * 32 + lane) * 16);

  float accr[2][8];
  for (int i = 0; i < 2; i++)
    for (int r = 0; r < 8; r++) accr[i][r] = 0.0f;

  for (int k = 0; k < kmax; k++) {
    // stage x_t rows via async global->LDS (16B/lane x 4)
    for (int q = 0; q < 4; q++) {
      int fo  = (tid + 128 * q) * 4;
      int m   = fo >> 7, col = fo & 127;
      int len = s_len[m];
      int t   = (k < len - 1) ? k : (len - 1);
      const float* g = seqs + (size_t)(s_start[m] + t) * Idim + col;
      async_ld_b128(&s_xrow[fo], g);
    }
    { // stage h_b(t) rows via async (8 f16 = 16B per thread)
      int m  = tid >> 3;
      int j0 = (tid & 7) * 8;
      int len = s_len[m];
      int t   = (k < len - 1) ? k : (len - 1);
      const f16* g = hb + ((size_t)(b0 + m) * Tmax + t) * Hdim + j0;
      async_ld_b128(&s_hbrow[m * Hdim + j0], g);
    }
    { // prefetch next step's x rows
      int m = tid >> 3;
      int len = s_len[m];
      int t2 = (k + 1 < len - 1) ? (k + 1) : (len - 1);
      __builtin_prefetch(seqs + (size_t)(s_start[m] + t2) * Idim + (tid & 7) * 16, 0, 0);
    }
    async_wait0();
    __syncthreads();
    for (int r = 0; r < 16; r++) { // pack x -> A fragments
      int idx = tid + 128 * r;
      int m = idx >> 7, j = idx & 127;
      s_xfrag[(j >> 5) * 512 + amap(m, j & 31)] = (f16)s_xrow[idx];
    }
    __syncthreads();
    // GEMMs: gx (K=128) + gh (K=64); wave owns 3 of 12 N-tiles
    v16h ax[4], ah[2];
    for (int c = 0; c < 4; c++) ax[c] = *(const v16h*)(s_xfrag + c * 512 + lane * 16);
    for (int c = 0; c < 2; c++) ah[c] = *(const v16h*)(s_hfrag + c * 512 + lane * 16);
    for (int i = 0; i < 3; i++) {
      int tg = 3 * wave + i;
      v8f cg = {}, ch = {};
      for (int c = 0; c < 4; c++) cg = wmma_f16(ax[c], wihr[i][c], cg);
      for (int c = 0; c < 2; c++) ch = wmma_f16(ah[c], whf[i][c], ch);
      int n = tg * 16 + (lane & 15);
      int hh = lane >> 4;
      for (int r = 0; r < 8; r++) {
        int m = r + 8 * hh;
        s_agx[m * Gdim + n] = cg[r];
        s_agh[m * Gdim + n] = ch[r];
      }
    }
    __syncthreads();
    // GRU update + pack [hf|hb] fragments for the MLP GEMM
    for (int r = 0; r < 8; r++) {
      int idx = tid + 128 * r;
      int m = idx >> 6, j = idx & 63;
      float ar  = (s_agx[m*Gdim + j]      + s_bih[j])      + (s_agh[m*Gdim + j]      + s_bhh[j]);
      float az  = (s_agx[m*Gdim + 64 + j] + s_bih[64 + j]) + (s_agh[m*Gdim + 64 + j] + s_bhh[64 + j]);
      float gxn =  s_agx[m*Gdim + 128 + j] + s_bih[128 + j];
      float ghn =  s_agh[m*Gdim + 128 + j] + s_bhh[128 + j];
      float rg = sigmoidf_(ar);
      float zg = sigmoidf_(az);
      float ng = tanhf(gxn + rg * ghn);
      float hold = s_hcur[idx];
      float hv = (k < s_len[m]) ? ((1.0f - zg) * ng + zg * hold) : hold;
      s_hcur[idx] = hv;
      s_hfrag[(j >> 5) * 512 + amap(m, j & 31)] = (f16)hv;
      s_mfrag[(j >> 5) * 512 + amap(m, j & 31)] = (f16)hv;
      int j2 = 64 + j;
      s_mfrag[(j2 >> 5) * 512 + amap(m, j2 & 31)] = s_hbrow[idx];
    }
    __syncthreads();
    // MLP: o = tanh([hf|hb] * WmT + bm); wave owns 2 of 8 N-tiles
    float o_reg[2][8];
    v16h am[4];
    for (int c = 0; c < 4; c++) am[c] = *(const v16h*)(s_mfrag + c * 512 + lane * 16);
    for (int i = 0; i < 2; i++) {
      v8f cm = {};
      for (int c = 0; c < 4; c++) cm = wmma_f16(am[c], wmf[i][c], cm);
      int n = (2 * wave + i) * 16 + (lane & 15);
      float bmn = s_bm[n];
      for (int r = 0; r < 8; r++) o_reg[i][r] = tanhf(cm[r] + bmn);
    }
    { // score s_m = o_m . ctx  (reduce across 16 lanes of each half)
      float c0 = s_ctx[(2 * wave) * 16 + (lane & 15)];
      float c1 = s_ctx[(2 * wave + 1) * 16 + (lane & 15)];
      for (int r = 0; r < 8; r++) {
        float p = o_reg[0][r] * c0 + o_reg[1][r] * c1;
        p += __shfl_xor(p, 1, 32);
        p += __shfl_xor(p, 2, 32);
        p += __shfl_xor(p, 4, 32);
        p += __shfl_xor(p, 8, 32);
        if ((lane & 15) == 0) atomicAdd(&s_score[r + 8 * (lane >> 4)], p);
      }
    }
    __syncthreads();
    // online softmax scalars per sequence
    if (tid < 16) {
      int m = tid;
      float s = s_score[m];
      s_score[m] = 0.0f;
      if (k < s_len[m]) {
        float mx  = s_maxv[m];
        float nmx = fmaxf(mx, s);
        float al  = __expf(mx - nmx);
        float wgt = __expf(s - nmx);
        s_maxv[m]  = nmx;
        s_sumv[m]  = s_sumv[m] * al + wgt;
        s_alpha[m] = al;
        s_w[m]     = wgt;
      } else {
        s_alpha[m] = 1.0f;
        s_w[m]     = 0.0f;
      }
    }
    __syncthreads();
    // acc = acc*alpha + w*o (registers)
    for (int i = 0; i < 2; i++)
      for (int r = 0; r < 8; r++) {
        int m = r + 8 * (lane >> 4);
        accr[i][r] = accr[i][r] * s_alpha[m] + s_w[m] * o_reg[i][r];
      }
  }

  for (int i = 0; i < 2; i++) {
    int n = (2 * wave + i) * 16 + (lane & 15);
    for (int r = 0; r < 8; r++) {
      int m = r + 8 * (lane >> 4);
      out[(size_t)(b0 + m) * Odim + n] = accr[i][r] / fmaxf(s_sumv[m], 1e-30f);
    }
  }
}

// ---------------------------------------------------------------------------
extern "C" void kernel_launch(void* const* d_in, const int* in_sizes, int n_in,
                              void* d_out, int out_size, void* d_ws, size_t ws_size,
                              hipStream_t stream) {
  const float* seqs  = (const float*)d_in[0];
  const int*   lens  = (const int*)d_in[1];
  const float* Wih_f = (const float*)d_in[2];
  const float* Whh_f = (const float*)d_in[3];
  const float* bih_f = (const float*)d_in[4];
  const float* bhh_f = (const float*)d_in[5];
  const float* Wih_b = (const float*)d_in[6];
  const float* Whh_b = (const float*)d_in[7];
  const float* bih_b = (const float*)d_in[8];
  const float* bhh_b = (const float*)d_in[9];
  const float* Wm    = (const float*)d_in[10];
  const float* bm    = (const float*)d_in[11];
  const float* ctx   = (const float*)d_in[12];
  float* out = (float*)d_out;

  // workspace layout (~33.8 MB total; hb is L2-resident on MI455X)
  char* w = (char*)d_ws;
  size_t off = 0;
  int* starts = (int*)(w + off);            off += 4096;
  f16* wihf   = (f16*)(w + off);            off += (size_t)Gdim * Idim * 2;
  f16* whhf   = (f16*)(w + off);            off += (size_t)Gdim * Hdim * 2;
  f16* wihb   = (f16*)(w + off);            off += (size_t)Gdim * Idim * 2;
  f16* whhb   = (f16*)(w + off);            off += (size_t)Gdim * Hdim * 2;
  f16* wmp    = (f16*)(w + off);            off += (size_t)Odim * Odim * 2;
  f16* hbuf   = (f16*)(w + off);            off += (size_t)Bsz * Tmax * Hdim * 2;

  k_starts<<<1, 32, 0, stream>>>(lens, starts);
  k_pack<<<(Gdim * Idim + 255) / 256, 256, 0, stream>>>(Wih_f, wihf, Gdim, Idim);
  k_pack<<<(Gdim * Hdim + 255) / 256, 256, 0, stream>>>(Whh_f, whhf, Gdim, Hdim);
  k_pack<<<(Gdim * Idim + 255) / 256, 256, 0, stream>>>(Wih_b, wihb, Gdim, Idim);
  k_pack<<<(Gdim * Hdim + 255) / 256, 256, 0, stream>>>(Whh_b, whhb, Gdim, Hdim);
  k_pack<<<(Odim * Odim + 255) / 256, 256, 0, stream>>>(Wm, wmp, Odim, Odim);

  k_bwd<<<Bsz / 16, 128, 0, stream>>>(seqs, starts, lens, wihb, whhb,
                                      bih_b, bhh_b, hbuf);
  k_fwd<<<Bsz / 16, 128, 0, stream>>>(seqs, starts, lens, wihf, whhf, wmp,
                                      bih_f, bhh_f, bm, ctx, hbuf, out);
}
